// SharedRegressionModel_65481071404378
// MI455X (gfx1250) — compile-verified, matrix operand order checked by reading, not run on hardware
//
#include <hip/hip_runtime.h>
#include <hip/hip_bf16.h>
#include <math.h>

#define NW   12
#define NL   3
#define DIM  4096          // 2^NW
#define TPB  256
#define AMPS (DIM / TPB)   // 16 amplitudes per thread (index bits 3..0)

typedef __attribute__((ext_vector_type(16))) _Float16 v16h;
typedef __attribute__((ext_vector_type(8)))  float    v8f;

// ---------------------------------------------------------------------------
// Kernel 1: register-resident state-vector simulation.
// One sample per 256-thread block; each thread holds 16 complex amplitudes in
// VGPRs. Wires 8-11 = in-register butterflies; wires 3-7 = wave32 shfl_xor;
// wires 0-2 = LDS exchange via ds_store_b128/ds_load_b128 (interleaved
// float2 complex, 16B-aligned); CNOT chain = one fused ds_load_b64 gather.
// Rotations on distinct wires commute, so reordering the gate block is exact.
// ---------------------------------------------------------------------------
__global__ __launch_bounds__(TPB) void qsim_kernel(const float* __restrict__ state_real,
                                                   const float* __restrict__ state_imag,
                                                   const float* __restrict__ weights,
                                                   float* __restrict__ qbuf) {
    __shared__ __align__(16) float2 cpx[DIM];   // interleaved (re,im): 32 KB
    __shared__ float gU[NL][NW][8];    // fused U = RY*RX: row0 = (U00,U01), row1 = (U10,U11)
    __shared__ float in0[NW][2];       // initial product state: column 0 of U_i
    __shared__ float in1[NW][2];
    __shared__ float zsum[NW];

    const int t    = threadIdx.x;
    const int b    = blockIdx.x;
    const int lane = t & 31;

    // --- fused layer gate matrices (weights are sample-independent) ---------
    if (t < NL * NW) {
        const int l = t / NW, i = t % NW;
        const float ax = weights[(l * NW + i) * 2 + 0] * 0.5f;
        const float ay = weights[(l * NW + i) * 2 + 1] * 0.5f;
        const float cx = cosf(ax), sx = sinf(ax);
        const float cy = cosf(ay), sy = sinf(ay);
        gU[l][i][0] =  cy * cx;  gU[l][i][1] =  sy * sx;   // U00
        gU[l][i][2] = -sy * cx;  gU[l][i][3] = -cy * sx;   // U01
        gU[l][i][4] =  sy * cx;  gU[l][i][5] = -cy * sx;   // U10
        gU[l][i][6] =  cy * cx;  gU[l][i][7] = -sy * sx;   // U11
    }
    // --- per-sample encoding gates: only column 0 matters (acts on |0>) -----
    if (t < NW) {
        const float xr = state_real[(size_t)b * DIM + t];
        const float xi = state_imag[(size_t)b * DIM + t];
        const float a  = atan2f(xi, xr) * 0.5f;
        const float c  = cosf(a), s = sinf(a);
        in0[t][0] = c * c;  in0[t][1] = s * s;      // U00
        in1[t][0] = s * c;  in1[t][1] = -c * s;     // U10
        zsum[t] = 0.0f;
    }
    __syncthreads();

    // --- initial state = product state, straight into registers -------------
    float ar[AMPS], ai[AMPS];
    #pragma unroll
    for (int j = 0; j < AMPS; ++j) {
        const int k = t * AMPS + j;
        float pr = 1.0f, pi = 0.0f;
        #pragma unroll
        for (int w = 0; w < NW; ++w) {
            const int bit = (k >> (NW - 1 - w)) & 1;    // wire w = index bit 11-w
            const float ur = bit ? in1[w][0] : in0[w][0];
            const float ui = bit ? in1[w][1] : in0[w][1];
            const float nr = pr * ur - pi * ui;
            const float ni = pr * ui + pi * ur;
            pr = nr; pi = ni;
        }
        ar[j] = pr; ai[j] = pi;
    }
    __syncthreads();   // in0/in1 no longer needed; cpx free for exchanges

    float4* const myslot = (float4*)(&cpx[t * AMPS]);   // 128B per thread, 16B aligned

    for (int l = 0; l < NL; ++l) {
        // ---- wires 8..11 (index bits 3..0): pure in-register butterflies ----
        #pragma unroll
        for (int w = 8; w < NW; ++w) {
            const int p = NW - 1 - w;                   // 3..0
            const int m = 1 << p;
            const float u00r = gU[l][w][0], u00i = gU[l][w][1];
            const float u01r = gU[l][w][2], u01i = gU[l][w][3];
            const float u10r = gU[l][w][4], u10i = gU[l][w][5];
            const float u11r = gU[l][w][6], u11i = gU[l][w][7];
            #pragma unroll
            for (int j = 0; j < AMPS; ++j) {
                if (j & m) continue;
                const int j1 = j | m;
                const float a0r = ar[j],  a0i = ai[j];
                const float a1r = ar[j1], a1i = ai[j1];
                ar[j]  = u00r * a0r - u00i * a0i + u01r * a1r - u01i * a1i;
                ai[j]  = u00r * a0i + u00i * a0r + u01r * a1i + u01i * a1r;
                ar[j1] = u10r * a0r - u10i * a0i + u11r * a1r - u11i * a1i;
                ai[j1] = u10r * a0i + u10i * a0r + u11r * a1i + u11i * a1r;
            }
        }
        // ---- wires 3..7 (index bits 8..4 = lane bits): shfl_xor exchange ----
        #pragma unroll
        for (int w = 3; w < 8; ++w) {
            const int p   = NW - 1 - w;                 // 8..4
            const int lm  = 1 << (p - 4);               // lane mask 16..1
            const int bit = (lane >> (p - 4)) & 1;
            const float u0r = gU[l][w][bit * 4 + 0], u0i = gU[l][w][bit * 4 + 1];
            const float u1r = gU[l][w][bit * 4 + 2], u1i = gU[l][w][bit * 4 + 3];
            #pragma unroll
            for (int j = 0; j < AMPS; ++j) {
                const float prr = __shfl_xor(ar[j], lm, 32);
                const float pii = __shfl_xor(ai[j], lm, 32);
                const float s0r = bit ? prr : ar[j], s0i = bit ? pii : ai[j];
                const float s1r = bit ? ar[j] : prr, s1i = bit ? ai[j] : pii;
                ar[j] = u0r * s0r - u0i * s0i + u1r * s1r - u1i * s1i;
                ai[j] = u0r * s0i + u0i * s0r + u1r * s1i + u1i * s1r;
            }
        }
        // ---- wires 0..2 (index bits 11..9): cross-wave LDS exchange ---------
        // state out as 8x ds_store_b128, partner in as 8x ds_load_b128
        #pragma unroll
        for (int w = 0; w < 3; ++w) {
            const int p   = NW - 1 - w;                 // 11..9
            const int tm  = 1 << (p - 4);               // thread mask 128..32
            const int bit = (t >> (p - 4)) & 1;
            const float u0r = gU[l][w][bit * 4 + 0], u0i = gU[l][w][bit * 4 + 1];
            const float u1r = gU[l][w][bit * 4 + 2], u1i = gU[l][w][bit * 4 + 3];

            auto bfly = [&](int j, float prr, float pii) {
                const float s0r = bit ? prr : ar[j], s0i = bit ? pii : ai[j];
                const float s1r = bit ? ar[j] : prr, s1i = bit ? ai[j] : pii;
                ar[j] = u0r * s0r - u0i * s0i + u1r * s1r - u1i * s1i;
                ai[j] = u0r * s0i + u0i * s0r + u1r * s1i + u1i * s1r;
            };

            #pragma unroll
            for (int h = 0; h < AMPS / 2; ++h)
                myslot[h] = make_float4(ar[2 * h], ai[2 * h], ar[2 * h + 1], ai[2 * h + 1]);
            __syncthreads();
            const float4* const ps = (const float4*)(&cpx[(t ^ tm) * AMPS]);
            #pragma unroll
            for (int h = 0; h < AMPS / 2; ++h) {
                const float4 pv = ps[h];
                bfly(2 * h,     pv.x, pv.y);
                bfly(2 * h + 1, pv.z, pv.w);
            }
            __syncthreads();
        }
        // ---- CNOT(0,1)..CNOT(10,11),CNOT(11,0) as one fused gather ----------
        #pragma unroll
        for (int h = 0; h < AMPS / 2; ++h)
            myslot[h] = make_float4(ar[2 * h], ai[2 * h], ar[2 * h + 1], ai[2 * h + 1]);
        __syncthreads();
        #pragma unroll
        for (int j = 0; j < AMPS; ++j) {
            const int y = t * AMPS + j;
            int x = y;
            x ^= ((x >> 0) & 1) << 11;                  // CNOT(11,0): bit11 ^= bit0
            #pragma unroll
            for (int c = 10; c >= 0; --c)               // CNOT(c,c+1), reverse order
                x ^= ((x >> (11 - c)) & 1) << (10 - c);
            const float2 v = cpx[x];                    // ds_load_b64
            ar[j] = v.x; ai[j] = v.y;
        }
        __syncthreads();
    }

    // --- <Z_i> directly from registers; wave32 shuffle reduce + LDS atomics -
    float zac[NW];
    #pragma unroll
    for (int i = 0; i < NW; ++i) zac[i] = 0.0f;
    #pragma unroll
    for (int j = 0; j < AMPS; ++j) {
        const int k = t * AMPS + j;
        const float p = ar[j] * ar[j] + ai[j] * ai[j];
        #pragma unroll
        for (int i = 0; i < NW; ++i)
            zac[i] += ((k >> (NW - 1 - i)) & 1) ? -p : p;
    }
    #pragma unroll
    for (int off = 16; off > 0; off >>= 1)
        #pragma unroll
        for (int i = 0; i < NW; ++i)
            zac[i] += __shfl_down(zac[i], off, 32);
    if (lane == 0)
        for (int i = 0; i < NW; ++i)
            atomicAdd(&zsum[i], zac[i]);            // ds_add_f32
    __syncthreads();

    if (t < 16)                                      // pad K=12..15 with zeros
        qbuf[(size_t)b * 16 + t] = (t < NW) ? zsum[t] : 0.0f;
}

// ---------------------------------------------------------------------------
// Kernel 2: head GEMV out = q @ head_w^T + b via v_wmma_f32_16x16x32_f16.
// One wave per 16-sample tile. B broadcasts head_w across all 16 columns, so
// every column of D holds the result; lanes 0 and 16 own column N=0.
// 16-bit A layout (wave32): lane<16 row M=lane holds K={0..7,16..23};
// lane>=16 row M=lane-16 holds K={8..15,24..31}. K>=12 zero-padded.
// ---------------------------------------------------------------------------
__global__ __launch_bounds__(32) void head_wmma_kernel(const float* __restrict__ qbuf,
                                                       const float* __restrict__ head_w,
                                                       const float* __restrict__ head_b,
                                                       float* __restrict__ out) {
    const int tile  = blockIdx.x;
    const int lane  = threadIdx.x;
    const int row   = lane & 15;
    const int upper = lane >> 4;

    v16h a, bm;
    #pragma unroll
    for (int e = 0; e < 16; ++e) {
        const int K = (e < 8 ? e : 8 + e) + (upper ? 8 : 0);
        const float av = (K < NW) ? qbuf[(size_t)(tile * 16 + row) * 16 + K] : 0.0f;
        const float bv = (K < NW) ? head_w[K] : 0.0f;
        a[e]  = (_Float16)av;
        bm[e] = (_Float16)bv;
    }
    v8f c = {};
    c = __builtin_amdgcn_wmma_f32_16x16x32_f16(false, a, false, bm,
                                               (short)0, c, false, false);
    const float bias = head_b[0];
    // C/D layout: VGPR r, lanes 0-15 -> (M=r, N=lane); lanes 16-31 -> (M=r+8).
    if (lane == 0) {
        #pragma unroll
        for (int r = 0; r < 8; ++r) out[tile * 16 + r] = c[r] + bias;
    } else if (lane == 16) {
        #pragma unroll
        for (int r = 0; r < 8; ++r) out[tile * 16 + 8 + r] = c[r] + bias;
    }
}

// ---------------------------------------------------------------------------
extern "C" void kernel_launch(void* const* d_in, const int* in_sizes, int n_in,
                              void* d_out, int out_size, void* d_ws, size_t ws_size,
                              hipStream_t stream) {
    const float* state_real = (const float*)d_in[0];
    const float* state_imag = (const float*)d_in[1];
    const float* weights    = (const float*)d_in[2];
    const float* head_w     = (const float*)d_in[3];
    const float* head_b     = (const float*)d_in[4];
    float* out  = (float*)d_out;
    float* qbuf = (float*)d_ws;                      // BATCH x 16 floats = 512 KB

    const int batch = in_sizes[0] / DIM;             // 8192

    qsim_kernel<<<batch, TPB, 0, stream>>>(state_real, state_imag, weights, qbuf);
    head_wmma_kernel<<<batch / 16, 32, 0, stream>>>(qbuf, head_w, head_b, out);
}
